// GRU_91207925498430
// MI455X (gfx1250) — matrix operand order repaired and checked by reading, not compiled, not run on hardware
//
#include <hip/hip_runtime.h>
#include <math.h>

// ---------------------------------------------------------------------------
// GRU LM for MI455X (gfx1250, wave32, WMMA f16 16x16x32, f32 accumulate)
//
// Phase 1: convert weights f32->f16 (L2-resident afterwards)
// Phase 2: embedding gather * sqrt(512) -> f16
// Phase 3: persistent single-workgroup GRU recurrence (LDS-resident state)
// Phase 4: one big 4096x32000x512 WMMA GEMM for logits (dominant FLOPs)
// ---------------------------------------------------------------------------

typedef _Float16 v16h __attribute__((ext_vector_type(16)));
typedef _Float16 v8h  __attribute__((ext_vector_type(8)));
typedef float    v8f  __attribute__((ext_vector_type(8)));

#define SEQ    128
#define BATCH  32
#define VOCAB  32000
#define HID    512
#define CAT    1024          // HID + EMB
#define NLOGIT ((size_t)SEQ * BATCH * VOCAB)   // 131,072,000

__device__ __forceinline__ v8f wmma_f16(v16h a, v16h b, v8f c) {
    // 8 args: (neg_a, A, neg_b, B, c_mod, C, reuse_a, reuse_b)
    return __builtin_amdgcn_wmma_f32_16x16x32_f16(false, a, false, b,
                                                  (short)0, c, false, false);
}

// A-matrix fragment (16x32 f16, MxK). ISA layout: lane = half*16 + ml,
// VGPR0..3 hold K = half*8 + {0..7}, VGPR4..7 hold K = 16 + half*8 + {0..7}.
// Both runs are 16B-contiguous -> two b128 loads per lane.
__device__ __forceinline__ v16h load_fragA(const _Float16* base, int stride,
                                           int m0, int k0, int lane) {
    const int half = lane >> 4;
    const int ml   = lane & 15;
    const _Float16* p = base + (size_t)(m0 + ml) * stride + k0 + half * 8;
    v8h lo = *(const v8h*)(p);        // K = k0+half*8 .. +7
    v8h hi = *(const v8h*)(p + 16);   // K = k0+16+half*8 .. +7
    return __builtin_shufflevector(lo, hi, 0,1,2,3,4,5,6,7,8,9,10,11,12,13,14,15);
}

// B-matrix fragment (32x16 f16, KxN) for C = A @ W^T, B[k][n] = W[n][k].
// Same K order as A: VGPR0..3 hold K = half*8 + {0..7}, VGPR4..7 hold
// K = 16 + half*8 + {0..7} of column n -> two b128 from W row n.
__device__ __forceinline__ v16h load_fragB(const _Float16* w, int stride,
                                           int n0, int k0, int lane) {
    const int half = lane >> 4;
    const int nl   = lane & 15;
    const _Float16* p = w + (size_t)(n0 + nl) * stride + k0 + half * 8;
    v8h lo = *(const v8h*)(p);
    v8h hi = *(const v8h*)(p + 16);
    return __builtin_shufflevector(lo, hi, 0,1,2,3,4,5,6,7,8,9,10,11,12,13,14,15);
}

__device__ __forceinline__ float sigmoidf_(float x) {
    return 1.0f / (1.0f + __expf(-x));
}

// --------------------------- weight conversion -----------------------------
__global__ void cvt_f32_to_f16(const float* __restrict__ src,
                               _Float16* __restrict__ dst, int n) {
    int i = blockIdx.x * blockDim.x + threadIdx.x;
    int stride = gridDim.x * blockDim.x;
    for (; i < n; i += stride) dst[i] = (_Float16)src[i];
}

// --------------------------- embedding gather ------------------------------
__global__ void embed_kernel(const int* __restrict__ tokens,
                             const float* __restrict__ emb,
                             _Float16* __restrict__ x0) {
    const int row = blockIdx.x;             // t*BATCH + b, 0..4095
    const int tok = tokens[row];
    const float scale = 22.62741699796952f; // sqrt(512)
    for (int e = threadIdx.x; e < HID; e += blockDim.x)
        x0[(size_t)row * HID + e] = (_Float16)(emb[(size_t)tok * HID + e] * scale);
}

// ------------------------ persistent GRU recurrence ------------------------
// One workgroup (16 waves). LDS: h[2][32][512] + x[32][512] + rh[32][512] f16
// = 128 KB (of 320 KB/WGP). Wave w owns n-columns [32w, 32w+32) = two n-tiles,
// and both m-tiles (batch 32 = 2 x 16). z-gate stays in registers.
__global__ __launch_bounds__(512)
void gru_kernel(const _Float16* __restrict__ x0,     // [SEQ*BATCH, HID]
                const _Float16* __restrict__ Wr16,   // [2, HID, CAT]
                const _Float16* __restrict__ Wz16,
                const _Float16* __restrict__ Wh16,
                const float* __restrict__ br,        // [2, HID]
                const float* __restrict__ bz,
                const float* __restrict__ bh,
                const float* __restrict__ h0,        // [2, BATCH, HID]
                _Float16* __restrict__ x2,           // [SEQ*BATCH, HID]
                float* __restrict__ hfinal)          // [2*BATCH*HID]
{
    extern __shared__ _Float16 smem[];
    _Float16* sh  = smem;                       // [2][BATCH][HID] hidden state
    _Float16* sx  = smem + 2 * BATCH * HID;     // [BATCH][HID] layer input
    _Float16* srh = sx + BATCH * HID;           // [BATCH][HID] r * h

    const int tid  = threadIdx.x;
    const int wave = tid >> 5;      // 0..15
    const int lane = tid & 31;
    const int half = lane >> 4;
    const int nl   = lane & 15;
    const int n0   = wave * 32;     // this wave's n-tile base

    for (int i = tid; i < 2 * BATCH * HID; i += 512)
        sh[i] = (_Float16)h0[i];
    __syncthreads();

    for (int t = 0; t < SEQ; ++t) {
        for (int i = tid; i < BATCH * HID; i += 512)
            sx[i] = x0[(size_t)t * BATCH * HID + i];
        __syncthreads();

        for (int l = 0; l < 2; ++l) {
            const _Float16* wr = Wr16 + (size_t)l * HID * CAT;
            const _Float16* wz = Wz16 + (size_t)l * HID * CAT;
            const _Float16* wh = Wh16 + (size_t)l * HID * CAT;
            _Float16* hl = sh + l * BATCH * HID;

            // ---- Phase A: r and z gates (hx = [h | x]) ----
            v8f accR[2][2] = {{{0.f},{0.f}},{{0.f},{0.f}}};
            v8f accZ[2][2] = {{{0.f},{0.f}},{{0.f},{0.f}}};
            for (int k0 = 0; k0 < CAT; k0 += 32) {
                const _Float16* Asrc = (k0 < HID) ? hl : sx;
                const int kk = (k0 < HID) ? k0 : (k0 - HID);
                v16h a0 = load_fragA(Asrc, HID, 0,  kk, lane);
                v16h a1 = load_fragA(Asrc, HID, 16, kk, lane);
                #pragma unroll
                for (int j = 0; j < 2; ++j) {
                    v16h bR = load_fragB(wr, CAT, n0 + 16 * j, k0, lane);
                    v16h bZ = load_fragB(wz, CAT, n0 + 16 * j, k0, lane);
                    accR[0][j] = wmma_f16(a0, bR, accR[0][j]);
                    accR[1][j] = wmma_f16(a1, bR, accR[1][j]);
                    accZ[0][j] = wmma_f16(a0, bZ, accZ[0][j]);
                    accZ[1][j] = wmma_f16(a1, bZ, accZ[1][j]);
                }
            }
            // activations: rt -> LDS (needed by all waves), zt stays in regs
            #pragma unroll
            for (int j = 0; j < 2; ++j) {
                const int n = n0 + 16 * j + nl;
                const float biasR = br[l * HID + n];
                const float biasZ = bz[l * HID + n];
                #pragma unroll
                for (int i = 0; i < 2; ++i) {
                    #pragma unroll
                    for (int r = 0; r < 8; ++r) {
                        const int m = 16 * i + (half ? r + 8 : r);
                        const float rt = sigmoidf_(accR[i][j][r] + biasR);
                        accZ[i][j][r]  = sigmoidf_(accZ[i][j][r] + biasZ);
                        srh[m * HID + n] =
                            (_Float16)(rt * (float)hl[m * HID + n]);
                    }
                }
            }
            __syncthreads();

            // ---- Phase B: candidate ht (rhx = [r*h | x]) ----
            v8f accH[2][2] = {{{0.f},{0.f}},{{0.f},{0.f}}};
            for (int k0 = 0; k0 < CAT; k0 += 32) {
                const _Float16* Asrc = (k0 < HID) ? srh : sx;
                const int kk = (k0 < HID) ? k0 : (k0 - HID);
                v16h a0 = load_fragA(Asrc, HID, 0,  kk, lane);
                v16h a1 = load_fragA(Asrc, HID, 16, kk, lane);
                #pragma unroll
                for (int j = 0; j < 2; ++j) {
                    v16h bH = load_fragB(wh, CAT, n0 + 16 * j, k0, lane);
                    accH[0][j] = wmma_f16(a0, bH, accH[0][j]);
                    accH[1][j] = wmma_f16(a1, bH, accH[1][j]);
                }
            }
            __syncthreads();   // all frag reads of sx/srh done before overwrite

            // ---- out = (1-z)*h + z*ht ; update state / feed next layer ----
            #pragma unroll
            for (int j = 0; j < 2; ++j) {
                const int n = n0 + 16 * j + nl;
                const float biasH = bh[l * HID + n];
                #pragma unroll
                for (int i = 0; i < 2; ++i) {
                    #pragma unroll
                    for (int r = 0; r < 8; ++r) {
                        const int m  = 16 * i + (half ? r + 8 : r);
                        const float ht = tanhf(accH[i][j][r] + biasH);
                        const float z  = accZ[i][j][r];
                        const float hv = (float)hl[m * HID + n];
                        const float o  = (1.0f - z) * hv + z * ht;
                        const _Float16 o16 = (_Float16)o;
                        hl[m * HID + n] = o16;
                        if (l == 0) sx[m * HID + n] = o16;
                        else x2[((size_t)t * BATCH + m) * HID + n] = o16;
                    }
                }
            }
            __syncthreads();
        }
    }

    for (int i = tid; i < 2 * BATCH * HID; i += 512)
        hfinal[i] = (float)sh[i];
}

// --------------------------- logits GEMM (dominant) ------------------------
// C[4096,32000] = X2[4096,512] @ fc_w^T + fc_b, f16 WMMA, f32 out.
// Wave: 4 m-tiles x 2 n-tiles (64x32). Block: 8 waves along N -> 64x256.
// Grid: (32000/256=125, 4096/64=64). All operands L2-resident.
__global__ __launch_bounds__(256)
void fc_kernel(const _Float16* __restrict__ X,     // [4096, 512]
               const _Float16* __restrict__ Wt,    // [32000, 512] (fc_w, f16)
               const float* __restrict__ bias,     // [32000]
               float* __restrict__ out)            // [4096, 32000]
{
    const int lane   = threadIdx.x & 31;
    const int wave   = threadIdx.x >> 5;
    const int m_base = blockIdx.y * 64;
    const int n_base = blockIdx.x * 256 + wave * 32;

    v8f acc[4][2] = {{{0.f},{0.f}},{{0.f},{0.f}},{{0.f},{0.f}},{{0.f},{0.f}}};

    for (int k0 = 0; k0 < HID; k0 += 32) {
        v16h a[4];
        #pragma unroll
        for (int i = 0; i < 4; ++i)
            a[i] = load_fragA(X, HID, m_base + 16 * i, k0, lane);
        #pragma unroll
        for (int j = 0; j < 2; ++j) {
            v16h b = load_fragB(Wt, HID, n_base + 16 * j, k0, lane);
            #pragma unroll
            for (int i = 0; i < 4; ++i)
                acc[i][j] = wmma_f16(a[i], b, acc[i][j]);
        }
        if (k0 + 32 < HID)  // pull next weight K-chunk toward the WGP
            __builtin_prefetch(Wt + (size_t)(n_base + (lane & 15)) * HID + k0 + 32, 0, 3);
    }

    const int half = lane >> 4;
    const int nl   = lane & 15;
    #pragma unroll
    for (int j = 0; j < 2; ++j) {
        const int n = n_base + 16 * j + nl;
        const float bv = bias[n];
        #pragma unroll
        for (int i = 0; i < 4; ++i) {
            #pragma unroll
            for (int r = 0; r < 8; ++r) {
                const int m = m_base + 16 * i + (half ? r + 8 : r);
                out[(size_t)m * VOCAB + n] = acc[i][j][r] + bv;
            }
        }
    }
}

// ---------------------------------------------------------------------------
extern "C" void kernel_launch(void* const* d_in, const int* in_sizes, int n_in,
                              void* d_out, int out_size, void* d_ws, size_t ws_size,
                              hipStream_t stream) {
    const int*   tokens = (const int*)  d_in[0];
    const float* emb    = (const float*)d_in[1];
    const float* Wr     = (const float*)d_in[2];
    const float* br     = (const float*)d_in[3];
    const float* Wz     = (const float*)d_in[4];
    const float* bz     = (const float*)d_in[5];
    const float* Wh     = (const float*)d_in[6];
    const float* bh     = (const float*)d_in[7];
    const float* fcw    = (const float*)d_in[8];
    const float* fcb    = (const float*)d_in[9];
    const float* h0     = (const float*)d_in[10];
    float* out = (float*)d_out;

    // workspace carve-up (~47 MB)
    char* ws = (char*)d_ws;
    size_t off = 0;
    const int NGATE = 2 * HID * CAT;             // 1,048,576 per gate matrix
    _Float16* Wr16  = (_Float16*)(ws + off); off += (size_t)NGATE * 2;
    _Float16* Wz16  = (_Float16*)(ws + off); off += (size_t)NGATE * 2;
    _Float16* Wh16  = (_Float16*)(ws + off); off += (size_t)NGATE * 2;
    _Float16* fcw16 = (_Float16*)(ws + off); off += (size_t)VOCAB * HID * 2;
    _Float16* x0_16 = (_Float16*)(ws + off); off += (size_t)SEQ * BATCH * HID * 2;
    _Float16* x2_16 = (_Float16*)(ws + off); off += (size_t)SEQ * BATCH * HID * 2;
    (void)off; (void)ws_size; (void)in_sizes; (void)n_in; (void)out_size;

    cvt_f32_to_f16<<<512, 256, 0, stream>>>(Wr, Wr16, NGATE);
    cvt_f32_to_f16<<<512, 256, 0, stream>>>(Wz, Wz16, NGATE);
    cvt_f32_to_f16<<<512, 256, 0, stream>>>(Wh, Wh16, NGATE);
    cvt_f32_to_f16<<<2048, 256, 0, stream>>>(fcw, fcw16, VOCAB * HID);

    embed_kernel<<<SEQ * BATCH, 256, 0, stream>>>(tokens, emb, x0_16);

    // persistent recurrence: 1 workgroup, 16 waves, 128 KB dynamic LDS
    gru_kernel<<<1, 512, 128 * 1024, stream>>>(x0_16, Wr16, Wz16, Wh16,
                                               br, bz, bh, h0, x2_16,
                                               out + NLOGIT);

    dim3 grid(VOCAB / 256, (SEQ * BATCH) / 64);  // (125, 64)
    fc_kernel<<<grid, 256, 0, stream>>>(x2_16, fcw16, fcb, out);
}
